// GATEncoder_32349693673735
// MI455X (gfx1250) — matrix operand order repaired and checked by reading, not compiled, not run on hardware
//
#include <hip/hip_runtime.h>
#include <hip/hip_bf16.h>
#include <stdint.h>

// GAT encoder: 2x GATConv on MI455X (gfx1250, wave32, WMMA).
// GEMMs: A-panel staged to LDS via global_load_async_to_lds_b128 (ASYNCcnt),
// fragments fed to v_wmma_f32_16x16x32_bf16. Message passing is a 3-pass
// atomic segment-softmax (bandwidth bound, ~0.9 GB/layer of edge messages).

#define IN_CH   128
#define HIDC    256
#define HEADS   4
#define C1HEAD  64
#define NEG_SLOPE 0.2f

typedef __attribute__((ext_vector_type(16))) __bf16 v16bf;
typedef __attribute__((ext_vector_type(8)))  __bf16 v8bf;
typedef __attribute__((ext_vector_type(8)))  float  v8f;
typedef __attribute__((ext_vector_type(4)))  float  v4f;

// ---------------------------------------------------------------- GEMM (WMMA)
// C[M,N] = A[M,K](f32, row-major) * Bt[N,K](bf16, row-major = B transposed)
// One wave -> one 16x16 f32 tile. Block = 256 thr = 8 waves = 32(M) x 64(N).
// A panel (32 x K f32) is staged through LDS with async-to-LDS copies.
template <int K, int N>
__global__ void __launch_bounds__(256)
gemm_wmma_bf16(const float* __restrict__ A, const __bf16* __restrict__ Bt,
               float* __restrict__ C, int M) {
    constexpr int LDS_STRIDE = K + 4;            // +4 dwords: bank-spread rows
    __shared__ __align__(16) float ldsA[32 * LDS_STRIDE];

    const int wave = threadIdx.x >> 5;
    const int lane = threadIdx.x & 31;
    const int wm = wave >> 2;            // 0..1
    const int wn = wave & 3;             // 0..3
    const int tileM = blockIdx.x * 32 + wm * 16;
    const int tileN = blockIdx.y * 64 + wn * 16;

    // ---- stage A panel (rows blockIdx.x*32 .. +31, all K cols) into LDS ----
    constexpr int K4 = K / 4;                    // 16B chunks per row
    const unsigned ldsBase = (unsigned)(uintptr_t)(&ldsA[0]);
#pragma unroll
    for (int ci = 0; ci < 32 * K4 / 256; ++ci) {
        int idx = ci * 256 + threadIdx.x;
        int r  = idx / K4;
        int c4 = idx - r * K4;
        int gr = blockIdx.x * 32 + r;
        if (gr >= M) gr = M - 1;                 // clamp; stores are guarded
        const float* gp = A + (size_t)gr * K + c4 * 4;
        unsigned lo = ldsBase + (unsigned)(r * LDS_STRIDE + c4 * 4) * 4u;
        asm volatile("global_load_async_to_lds_b128 %0, %1, off"
                     :: "v"(lo), "v"(gp) : "memory");
    }
    asm volatile("s_wait_asynccnt 0x0" ::: "memory");
    __syncthreads();

    // A fragment addressing (16-bit A 16x32 layout):
    //   lane L: row M = tileM + (L&15), K-base = (L>=16)?8:0,
    //   elements = {kb..kb+7, kb+16..kb+23}
    const int lr   = wm * 16 + (lane & 15);      // local row in LDS panel
    const int kbA  = (lane >> 4) << 3;           // 0 or 8
    const float* Arow = &ldsA[lr * LDS_STRIDE];

    // B fragment (16-bit B 32x16): lane L: col N = tileN + (L&15),
    //   K-base = (L>=16)?8:0, elements = {kb..kb+7, kb+16..kb+23}
    const int ncol = tileN + (lane & 15);
    const int kbB  = (lane >> 4) << 3;           // 0 or 8
    const __bf16* Brow = Bt + (size_t)ncol * K;  // N is a multiple of 64 here

    v8f acc = {};
#pragma unroll
    for (int k0 = 0; k0 < K; k0 += 32) {
        const float* ap = Arow + k0 + kbA;
        v4f a0 = *(const v4f*)(ap);
        v4f a1 = *(const v4f*)(ap + 4);
        v4f a2 = *(const v4f*)(ap + 16);
        v4f a3 = *(const v4f*)(ap + 20);
        v16bf af;
#pragma unroll
        for (int j = 0; j < 4; ++j) {
            af[j]      = (__bf16)a0[j];
            af[4 + j]  = (__bf16)a1[j];
            af[8 + j]  = (__bf16)a2[j];
            af[12 + j] = (__bf16)a3[j];
        }
        v8bf b0 = *(const v8bf*)(Brow + k0 + kbB);
        v8bf b1 = *(const v8bf*)(Brow + k0 + kbB + 16);
        v16bf bfr;
#pragma unroll
        for (int j = 0; j < 8; ++j) { bfr[j] = b0[j]; bfr[8 + j] = b1[j]; }
        acc = __builtin_amdgcn_wmma_f32_16x16x32_bf16(
                  false, af, false, bfr, (short)0, acc, false, false);
    }

    // C/D layout: VGPR r -> row tileM + r + ((lane>=16)?8:0), col tileN+(lane&15)
    const int cn = tileN + (lane & 15);
    const int rb = tileM + ((lane >> 4) << 3);
    float* Cp = C + (size_t)rb * N + cn;
    if (rb + 7 < M) {
#pragma unroll
        for (int r = 0; r < 8; ++r) Cp[r * N] = acc[r];
    } else {
#pragma unroll
        for (int r = 0; r < 8; ++r) if (rb + r < M) Cp[r * N] = acc[r];
    }
}

// -------------------------------------------------------------- small helpers
__global__ void transpose_to_bf16(const float* __restrict__ W,
                                  __bf16* __restrict__ Wt, int K, int N) {
    int idx = blockIdx.x * blockDim.x + threadIdx.x;
    if (idx >= K * N) return;
    int n = idx / K, k = idx - n * K;
    Wt[(size_t)n * K + k] = (__bf16)W[(size_t)k * N + n];
}

__global__ void fill_f32(float* __restrict__ p, float v, long long n) {
    long long i = (long long)blockIdx.x * blockDim.x + threadIdx.x;
    if (i < n) p[i] = v;
}
__global__ void fill_u32(unsigned* __restrict__ p, unsigned v, long long n) {
    long long i = (long long)blockIdx.x * blockDim.x + threadIdx.x;
    if (i < n) p[i] = v;
}

// per-node attention coefficients: as[n,h] = <h[n,h,:], a_src[h,:]>, same for ad
__global__ void node_alpha(const float* __restrict__ h,
                           const float* __restrict__ a_src,
                           const float* __restrict__ a_dst,
                           float* __restrict__ as, float* __restrict__ ad,
                           int nNodes, int H, int Cc) {
    int idx = blockIdx.x * blockDim.x + threadIdx.x;
    if (idx >= nNodes * H) return;
    int n = idx / H, hh = idx - n * H;
    const float* hp = h + ((size_t)n * H + hh) * Cc;
    const float* s0 = a_src + hh * Cc;
    const float* d0 = a_dst + hh * Cc;
    float s = 0.f, d = 0.f;
    for (int c = 0; c < Cc; ++c) { float v = hp[c]; s += v * s0[c]; d += v * d0[c]; }
    as[idx] = s; ad[idx] = d;
}

// float <-> monotone uint key (for atomicMax on signed floats)
__device__ __forceinline__ unsigned f2key(float f) {
    int i = __float_as_int(f);
    return (i >= 0) ? ((unsigned)i ^ 0x80000000u) : ~(unsigned)i;
}
__device__ __forceinline__ float key2f(unsigned k) {
    int i = (k & 0x80000000u) ? (int)(k ^ 0x80000000u) : (int)~k;
    return __int_as_float(i);
}
__device__ __forceinline__ void edge_nodes(const int* __restrict__ ei, int E,
                                           int e, int& s, int& d) {
    if (e < E) { s = ei[e]; d = ei[E + e]; }      // edge_index[0,e] / [1,e]
    else       { s = e - E; d = e - E; }          // appended self-loop
}

// pass A: segment max of leaky-relu'd logits
__global__ void edge_max_k(const int* __restrict__ ei, int E, int Etot,
                           const float* __restrict__ as, const float* __restrict__ ad,
                           unsigned* __restrict__ mkey, int H) {
    int idx = blockIdx.x * blockDim.x + threadIdx.x;
    if (idx >= Etot * H) return;
    int e = idx / H, hh = idx - e * H;
    int s, d; edge_nodes(ei, E, e, s, d);
    float ev = as[s * H + hh] + ad[d * H + hh];
    ev = ev > 0.f ? ev : ev * NEG_SLOPE;
    atomicMax(&mkey[d * H + hh], f2key(ev));
}

// pass B: segment sum of exp(e - max)
__global__ void edge_sum_k(const int* __restrict__ ei, int E, int Etot,
                           const float* __restrict__ as, const float* __restrict__ ad,
                           const unsigned* __restrict__ mkey,
                           float* __restrict__ denom, int H) {
    int idx = blockIdx.x * blockDim.x + threadIdx.x;
    if (idx >= Etot * H) return;
    int e = idx / H, hh = idx - e * H;
    int s, d; edge_nodes(ei, E, e, s, d);
    int dh = d * H + hh;
    float ev = as[s * H + hh] + ad[dh];
    ev = ev > 0.f ? ev : ev * NEG_SLOPE;
    atomicAdd(&denom[dh], __expf(ev - key2f(mkey[dh])));
}

// pass C: acc[d] += alpha * h[s]  (grid-stride over Etot*H*C elements)
__global__ void edge_agg_k(const int* __restrict__ ei, int E, int Etot,
                           const float* __restrict__ as, const float* __restrict__ ad,
                           const unsigned* __restrict__ mkey,
                           const float* __restrict__ denom,
                           const float* __restrict__ h, float* __restrict__ acc,
                           int H, int Cc) {
    long long total  = (long long)Etot * H * Cc;
    long long stride = (long long)gridDim.x * blockDim.x;
    for (long long idx = (long long)blockIdx.x * blockDim.x + threadIdx.x;
         idx < total; idx += stride) {
        int c = (int)(idx % Cc);
        long long t = idx / Cc;
        int hh = (int)(t % H);
        int e  = (int)(t / H);
        int s, d; edge_nodes(ei, E, e, s, d);
        int dh = d * H + hh;
        float ev = as[s * H + hh] + ad[dh];
        ev = ev > 0.f ? ev : ev * NEG_SLOPE;
        float w = __expf(ev - key2f(mkey[dh])) / (denom[dh] + 1e-16f);
        atomicAdd(&acc[((size_t)d * H + hh) * Cc + c],
                  h[((size_t)s * H + hh) * Cc + c] * w);
    }
}

__global__ void elu_bias_k(float* __restrict__ a, const float* __restrict__ b,
                           long long n, int cols) {
    long long i = (long long)blockIdx.x * blockDim.x + threadIdx.x;
    if (i >= n) return;
    float v = a[i] + b[i % cols];
    a[i] = v > 0.f ? v : expm1f(v);
}
__global__ void add_bias_k(float* __restrict__ a, const float* __restrict__ b,
                           long long n, int cols) {
    long long i = (long long)blockIdx.x * blockDim.x + threadIdx.x;
    if (i >= n) return;
    a[i] = a[i] + b[i % cols];
}

// ------------------------------------------------------------------- launcher
extern "C" void kernel_launch(void* const* d_in, const int* in_sizes, int n_in,
                              void* d_out, int out_size, void* d_ws, size_t ws_size,
                              hipStream_t stream) {
    const float* x    = (const float*)d_in[0];
    const int*   ei   = (const int*)d_in[1];
    const float* W1   = (const float*)d_in[2];
    const float* a_s1 = (const float*)d_in[3];
    const float* a_d1 = (const float*)d_in[4];
    const float* b1   = (const float*)d_in[5];
    const float* W2   = (const float*)d_in[6];
    const float* a_s2 = (const float*)d_in[7];
    const float* a_d2 = (const float*)d_in[8];
    const float* b2   = (const float*)d_in[9];
    float* out = (float*)d_out;
    (void)n_in; (void)out_size; (void)ws_size;

    const int N    = in_sizes[0] / IN_CH;
    const int E    = in_sizes[1] / 2;
    const int Etot = E + N;

    // workspace partition (256B aligned slices)
    char* p = (char*)d_ws;
    auto take = [&](size_t bytes) {
        char* r = p; p += (bytes + 255) & ~(size_t)255; return (void*)r;
    };
    float*    h_tmp = (float*)take((size_t)N * HIDC * 4);      // gemm outputs
    float*    accb  = (float*)take((size_t)N * HIDC * 4);      // aggregation
    float*    asN   = (float*)take((size_t)N * HEADS * 4);
    float*    adN   = (float*)take((size_t)N * HEADS * 4);
    unsigned* mkey  = (unsigned*)take((size_t)N * HEADS * 4);
    float*    denom = (float*)take((size_t)N * HEADS * 4);
    __bf16*   Wt1   = (__bf16*)take((size_t)IN_CH * HIDC * 2);
    __bf16*   Wt2   = (__bf16*)take((size_t)HIDC * HIDC * 2);

    const int T = 256;
    auto cdiv = [](long long a, long long b) { return (int)((a + b - 1) / b); };
    const unsigned NEG_INF_KEY = 0x007FFFFFu;   // f2key(-inf)

    // weights -> bf16, transposed (B^T rows = output columns, contiguous in K)
    transpose_to_bf16<<<cdiv((long long)IN_CH * HIDC, T), T, 0, stream>>>(W1, Wt1, IN_CH, HIDC);
    transpose_to_bf16<<<cdiv((long long)HIDC * HIDC, T), T, 0, stream>>>(W2, Wt2, HIDC, HIDC);

    dim3 gGemm(cdiv(N, 32), HIDC / 64);

    // ---------------- layer 1: GATConv(128 -> 64 x 4 heads, concat) ----------
    gemm_wmma_bf16<IN_CH, HIDC><<<gGemm, 256, 0, stream>>>(x, Wt1, h_tmp, N);
    node_alpha<<<cdiv((long long)N * HEADS, T), T, 0, stream>>>(
        h_tmp, a_s1, a_d1, asN, adN, N, HEADS, C1HEAD);
    fill_u32<<<cdiv((long long)N * HEADS, T), T, 0, stream>>>(mkey, NEG_INF_KEY, (long long)N * HEADS);
    fill_f32<<<cdiv((long long)N * HEADS, T), T, 0, stream>>>(denom, 0.f, (long long)N * HEADS);
    fill_f32<<<cdiv((long long)N * HIDC, T), T, 0, stream>>>(accb, 0.f, (long long)N * HIDC);
    edge_max_k<<<cdiv((long long)Etot * HEADS, T), T, 0, stream>>>(ei, E, Etot, asN, adN, mkey, HEADS);
    edge_sum_k<<<cdiv((long long)Etot * HEADS, T), T, 0, stream>>>(ei, E, Etot, asN, adN, mkey, denom, HEADS);
    edge_agg_k<<<8192, T, 0, stream>>>(ei, E, Etot, asN, adN, mkey, denom,
                                       h_tmp, accb, HEADS, C1HEAD);
    elu_bias_k<<<cdiv((long long)N * HIDC, T), T, 0, stream>>>(accb, b1, (long long)N * HIDC, HIDC);

    // ---------------- layer 2: GATConv(256 -> 256, 1 head) -------------------
    gemm_wmma_bf16<HIDC, HIDC><<<gGemm, 256, 0, stream>>>(accb, Wt2, h_tmp, N);
    node_alpha<<<cdiv((long long)N, T), T, 0, stream>>>(
        h_tmp, a_s2, a_d2, asN, adN, N, 1, HIDC);
    fill_u32<<<cdiv((long long)N, T), T, 0, stream>>>(mkey, NEG_INF_KEY, (long long)N);
    fill_f32<<<cdiv((long long)N, T), T, 0, stream>>>(denom, 0.f, (long long)N);
    fill_f32<<<cdiv((long long)N * HIDC, T), T, 0, stream>>>(out, 0.f, (long long)N * HIDC);
    edge_max_k<<<cdiv((long long)Etot, T), T, 0, stream>>>(ei, E, Etot, asN, adN, mkey, 1);
    edge_sum_k<<<cdiv((long long)Etot, T), T, 0, stream>>>(ei, E, Etot, asN, adN, mkey, denom, 1);
    edge_agg_k<<<8192, T, 0, stream>>>(ei, E, Etot, asN, adN, mkey, denom,
                                       h_tmp, out, 1, HIDC);
    add_bias_k<<<cdiv((long long)N * HIDC, T), T, 0, stream>>>(out, b2, (long long)N * HIDC, HIDC);
}
